// Attention_41695542509994
// MI455X (gfx1250) — matrix operand order repaired and checked by reading, not compiled, not run on hardware
//
#include <hip/hip_runtime.h>

#ifndef __has_builtin
#define __has_builtin(x) 0
#endif

typedef _Float16 f16;
typedef __attribute__((ext_vector_type(8)))  _Float16 h8;
typedef __attribute__((ext_vector_type(16))) _Float16 h16;
typedef __attribute__((ext_vector_type(8)))  float    f8;
typedef __attribute__((ext_vector_type(4)))  unsigned u32x4;
typedef __attribute__((ext_vector_type(8)))  unsigned u32x8;

#define DIM_B 2
#define DIM_N 2048
#define DIM_C 1024
#define DIM_H 16
#define DIM_D 64
#define M_TOT (DIM_B * DIM_N)   // 4096

// ---------------------------------------------------------------------------
// WMMA fragment helpers (CDNA5 ISA layouts, wave32)
// A (16xK f16): lane L<16 holds row L, K chunks {c..c+7, c+16..c+23}, c=0;
//               lane L>=16 holds row L-16 with c=8.
// B (Kx16 f16): lane L<16 holds col L, K chunks {c..c+7, c+16..c+23}, c=0;
//               lane L>=16 holds col L-16 with c=8.
// ---------------------------------------------------------------------------
static __device__ __forceinline__ h16 load_a_frag(const f16* row, int lhi) {
  const int c = lhi ? 8 : 0;
  h8 lo = *(const h8*)(row + c);
  h8 hi = *(const h8*)(row + c + 16);
  h16 r;
#pragma unroll
  for (int i = 0; i < 8; ++i) { r[i] = lo[i]; r[8 + i] = hi[i]; }
  return r;
}

static __device__ __forceinline__ h16 load_b_frag(const f16* row, int lhi) {
  const f16* p = row + (lhi ? 8 : 0);
  h8 lo = *(const h8*)(p);
  h8 hi = *(const h8*)(p + 16);
  h16 r;
#pragma unroll
  for (int i = 0; i < 8; ++i) { r[i] = lo[i]; r[8 + i] = hi[i]; }
  return r;
}

#define WMMA_F32_F16(a, b, c) \
  __builtin_amdgcn_wmma_f32_16x16x32_f16(false, (a), false, (b), (short)0, (c), false, false)

// ---------------------------------------------------------------------------
// fp32 -> fp16 conversion (x and weight matrices), 8 elems/thread
// ---------------------------------------------------------------------------
__global__ __launch_bounds__(256) void cvt_f32_to_f16(const float* __restrict__ in,
                                                      f16* __restrict__ out, int n) {
  int i = (blockIdx.x * 256 + threadIdx.x) * 8;
  if (i >= n) return;
  float4 a = *(const float4*)(in + i);
  float4 b = *(const float4*)(in + i + 4);
  h8 o;
  o[0] = (f16)a.x; o[1] = (f16)a.y; o[2] = (f16)a.z; o[3] = (f16)a.w;
  o[4] = (f16)b.x; o[5] = (f16)b.y; o[6] = (f16)b.z; o[7] = (f16)b.w;
  *(h8*)(out + i) = o;
}

// ---------------------------------------------------------------------------
// y = x @ W^T + b, output stored f16 in [B*H][N][D] layout.
// One wave -> 32x32 output tile (2 A-frags x 2 B-frags -> 4 WMMAs per K-step).
// ---------------------------------------------------------------------------
static __device__ __forceinline__ void store_tile(const f8& c, int m0, int n0,
                                                  const float* __restrict__ bias,
                                                  f16* __restrict__ out,
                                                  int lrow, int lhi) {
  const int col = n0 + lrow;
  const float bv = bias[col];
  const int h = col >> 6;        // head
  const int d = col & 63;        // dim within head
#pragma unroll
  for (int v = 0; v < 8; ++v) {
    int m  = m0 + lhi * 8 + v;
    int b_ = m >> 11;            // batch
    int n_ = m & (DIM_N - 1);    // token
    out[((size_t)(b_ * DIM_H + h) * DIM_N + n_) * DIM_D + d] = (f16)(c[v] + bv);
  }
}

__global__ __launch_bounds__(256) void qkv_gemm(const f16* __restrict__ x16,
                                                const f16* __restrict__ w16,
                                                const float* __restrict__ bias,
                                                f16* __restrict__ out) {
  const int lane = threadIdx.x & 31;
  const int wave = threadIdx.x >> 5;
  const int lrow = lane & 15;
  const int lhi  = lane >> 4;
  const int tile = blockIdx.x * 8 + wave;     // 4096 tiles total (128 x 32)
  const int nT   = DIM_C / 32;                // 32
  const int mt = tile / nT;
  const int nt = tile - mt * nT;
  const int m0 = mt * 32, n0 = nt * 32;

  const f16* ar0 = x16 + (size_t)(m0 + lrow) * DIM_C;
  const f16* ar1 = ar0 + (size_t)16 * DIM_C;
  const f16* br0 = w16 + (size_t)(n0 + lrow) * DIM_C;
  const f16* br1 = br0 + (size_t)16 * DIM_C;

  f8 c00 = {}, c01 = {}, c10 = {}, c11 = {};
#pragma unroll 2
  for (int k0 = 0; k0 < DIM_C; k0 += 32) {
    h16 a0 = load_a_frag(ar0 + k0, lhi);
    h16 a1 = load_a_frag(ar1 + k0, lhi);
    h16 b0 = load_b_frag(br0 + k0, lhi);
    h16 b1 = load_b_frag(br1 + k0, lhi);
    c00 = WMMA_F32_F16(a0, b0, c00);
    c01 = WMMA_F32_F16(a0, b1, c01);
    c10 = WMMA_F32_F16(a1, b0, c10);
    c11 = WMMA_F32_F16(a1, b1, c11);
  }

  store_tile(c00, m0,      n0,      bias, out, lrow, lhi);
  store_tile(c01, m0,      n0 + 16, bias, out, lrow, lhi);
  store_tile(c10, m0 + 16, n0,      bias, out, lrow, lhi);
  store_tile(c11, m0 + 16, n0 + 16, bias, out, lrow, lhi);
}

// ---------------------------------------------------------------------------
// Flash attention: block = one (b,h) x 128 query rows (8 waves x 16 rows).
// Double-buffered pipeline: while step t computes, the Tensor Data Mover pulls
// the K tile for step t+1 into the idle LDS buffer (hardware row padding gives
// the 72-f16 pitch) and V(t+1) global loads are in flight; V is scattered into
// the idle transposed buffer after compute. One barrier per step.
// ---------------------------------------------------------------------------
#define KROWS   32
#define KPITCH  72   // 64 + 8 pad (f16), 144B rows (16B aligned) == TDM pad
#define VPITCH  40   // 32 + 8 pad (f16), 80B rows  (16B aligned)
#define PPITCH  40
#define T_STEPS (DIM_N / KROWS)   // 64

__global__ __launch_bounds__(256) void attn_fwd(const f16* __restrict__ q16,
                                                const f16* __restrict__ k16,
                                                const f16* __restrict__ v16,
                                                float* __restrict__ out) {
  __shared__ f16 ldsK[2][KROWS * KPITCH];     // [buf][key][d] (TDM destination)
  __shared__ f16 ldsVt[2][DIM_D * VPITCH];    // [buf][d][key] (transposed)
  __shared__ f16 ldsP[8][16 * PPITCH];        // per-wave P staging [row][key]

  const int bh   = blockIdx.x >> 4;           // 0..31
  const int mblk = blockIdx.x & 15;
  const int wave = threadIdx.x >> 5;
  const int lane = threadIdx.x & 31;
  const int lrow = lane & 15;
  const int lhi  = lane >> 4;
  const int m0   = mblk * 128 + wave * 16;    // query-row tile base

  const f16* qbase = q16 + (size_t)bh * DIM_N * DIM_D;
  const f16* kbase = k16 + (size_t)bh * DIM_N * DIM_D;
  const f16* vbase = v16 + (size_t)bh * DIM_N * DIM_D;

  // Q fragments resident for the whole kernel (D=64 -> two K=32 A-fragments)
  const h16 aq0 = load_a_frag(qbase + (size_t)(m0 + lrow) * DIM_D + 0,  lhi);
  const h16 aq1 = load_a_frag(qbase + (size_t)(m0 + lrow) * DIM_D + 32, lhi);

  f8 o0 = {}, o1 = {}, o2 = {}, o3 = {};      // 16 x 64 output accumulator
  float rmax[8], rsum[8];
#pragma unroll
  for (int v = 0; v < 8; ++v) { rmax[v] = -1e30f; rsum[v] = 0.0f; }

  const int t_  = threadIdx.x;
  const int kr  = t_ >> 3;                    // key row 0..31
  const int kc  = (t_ & 7) * 8;               // d base 0..56

  f16* ldsPw = &ldsP[wave][0];
  const float SCALE = 0.03125f;               // 1/sqrt(1024)
  const float LOG2E = 1.442695041f;

  // TDM D# group1 (loop-invariant): data_size=1 (2B), pad_enable,
  // pad_interval=4 (32 DWORDs = one 64-f16 row), pad_amount=3 (4 DWORDs = 8 f16)
  // => LDS pitch 72 f16.  tensor=[64 x 32], tile=[64 x 32], dim0_stride=64.
  const u32x8 tdm_g1 = { (1u << 16) | (1u << 20) | (4u << 22) | (3u << 25),
                         64u << 16,          // tensor_dim0[15:0] in [63:48]
                         32u << 16,          // dim0[31:16]=0, tensor_dim1[15:0]
                         64u << 16,          // dim1[31:16]=0, tile_dim0=64
                         32u,                // tile_dim1=32, tile_dim2=0
                         64u,                // tensor_dim0_stride[31:0]=64
                         0u, 0u };
  const u32x4 tdm_gz = { 0u, 0u, 0u, 0u };
  const unsigned lds_k_addr0 = (unsigned)(uintptr_t)(&ldsK[0][0]);
  const unsigned lds_k_addr1 = (unsigned)(uintptr_t)(&ldsK[1][0]);

  auto tdm_issue = [&](int kv, unsigned ldsaddr) {
    unsigned long long ga =
        (unsigned long long)(uintptr_t)(kbase + (size_t)kv * DIM_D);
    u32x4 g0 = { 1u,                                    // count=1, user mode
                 ldsaddr,                               // lds_addr
                 (unsigned)ga,                          // global_addr[31:0]
                 (unsigned)((ga >> 32) & 0x1FFFFFFu) | (2u << 30) }; // type=2
    asm volatile("tensor_load_to_lds %0, %1, %2, %3"
                 :: "s"(g0), "s"(tdm_g1), "s"(tdm_gz), "s"(tdm_gz)
                 : "memory");
  };

  // ---- prologue: stage tile 0 into buffer 0 ----
  if (wave == 0) tdm_issue(0, lds_k_addr0);
  {
    h8 vreg = *(const h8*)(vbase + (size_t)kr * DIM_D + kc);
#pragma unroll
    for (int i = 0; i < 8; ++i) ldsVt[0][(kc + i) * VPITCH + kr] = vreg[i];
  }
  if (wave == 0) __builtin_amdgcn_s_wait_tensorcnt(0);
  __syncthreads();

  for (int t = 0; t < T_STEPS; ++t) {
    const int  cur  = t & 1;
    const int  nxt  = cur ^ 1;
    const bool more = (t + 1) < T_STEPS;
    const int  kvn  = (t + 1) * KROWS;

    // ---- kick off next tile: TDM for K, global loads for V (no waits) ----
    h8 vreg = {};
    if (more) {
      if (wave == 0) tdm_issue(kvn, nxt ? lds_k_addr1 : lds_k_addr0);
      vreg = *(const h8*)(vbase + (size_t)(kvn + kr) * DIM_D + kc);
    }

    const f16* Kc = &ldsK[cur][0];
    const f16* Vc = &ldsVt[cur][0];

    // ---- S = Q K^T for 32 keys: two 16x16 C tiles, K-dim = D = 64 ----
    f8 s0 = {}, s1 = {};
    {
      h16 b00 = load_b_frag(Kc + lrow * KPITCH + 0,  lhi);
      h16 b01 = load_b_frag(Kc + lrow * KPITCH + 32, lhi);
      s0 = WMMA_F32_F16(aq0, b00, s0);
      s0 = WMMA_F32_F16(aq1, b01, s0);
      h16 b10 = load_b_frag(Kc + (16 + lrow) * KPITCH + 0,  lhi);
      h16 b11 = load_b_frag(Kc + (16 + lrow) * KPITCH + 32, lhi);
      s1 = WMMA_F32_F16(aq0, b10, s1);
      s1 = WMMA_F32_F16(aq1, b11, s1);
    }

    // ---- online softmax: rows live in one VGPR across a 16-lane half ----
#pragma unroll
    for (int v = 0; v < 8; ++v) {
      float a = s0[v] * SCALE, b = s1[v] * SCALE;
      float mx = fmaxf(a, b);
      mx = fmaxf(mx, __shfl_xor(mx, 1, 32));
      mx = fmaxf(mx, __shfl_xor(mx, 2, 32));
      mx = fmaxf(mx, __shfl_xor(mx, 4, 32));
      mx = fmaxf(mx, __shfl_xor(mx, 8, 32));
      float mnew = fmaxf(rmax[v], mx);
      float corr = exp2f((rmax[v] - mnew) * LOG2E);
      float e0 = exp2f((a - mnew) * LOG2E);
      float e1 = exp2f((b - mnew) * LOG2E);
      float ls = e0 + e1;
      ls += __shfl_xor(ls, 1, 32);
      ls += __shfl_xor(ls, 2, 32);
      ls += __shfl_xor(ls, 4, 32);
      ls += __shfl_xor(ls, 8, 32);
      rmax[v] = mnew;
      rsum[v] = rsum[v] * corr + ls;
      o0[v] *= corr; o1[v] *= corr; o2[v] *= corr; o3[v] *= corr;
      // stage P (C-layout -> row-major LDS) for A-fragment reload
      int r = lhi * 8 + v;
      ldsPw[r * PPITCH + lrow]      = (f16)e0;
      ldsPw[r * PPITCH + 16 + lrow] = (f16)e1;
    }
#if __has_builtin(__builtin_amdgcn_s_wait_dscnt)
    __builtin_amdgcn_s_wait_dscnt(0);   // wave-local P RAW through LDS
#endif

    // ---- O += P V : A = P (16x32), B = V columns from transposed LDS ----
    {
      h16 ap  = load_a_frag(ldsPw + lrow * PPITCH, lhi);
      h16 bv0 = load_b_frag(Vc + (0  + lrow) * VPITCH, lhi);
      h16 bv1 = load_b_frag(Vc + (16 + lrow) * VPITCH, lhi);
      h16 bv2 = load_b_frag(Vc + (32 + lrow) * VPITCH, lhi);
      h16 bv3 = load_b_frag(Vc + (48 + lrow) * VPITCH, lhi);
      o0 = WMMA_F32_F16(ap, bv0, o0);
      o1 = WMMA_F32_F16(ap, bv1, o1);
      o2 = WMMA_F32_F16(ap, bv2, o2);
      o3 = WMMA_F32_F16(ap, bv3, o3);
    }

    // ---- finish next-tile staging, drain TDM, publish ----
    if (more) {
#pragma unroll
      for (int i = 0; i < 8; ++i) ldsVt[nxt][(kc + i) * VPITCH + kr] = vreg[i];
      if (wave == 0) __builtin_amdgcn_s_wait_tensorcnt(0);
    }
    __syncthreads();
  }

  // ---- normalize and store fp32 output [B][N][H*D] ----
  const int b_ = bh >> 4;
  const int h_ = bh & 15;
#pragma unroll
  for (int v = 0; v < 8; ++v) {
    float inv = 1.0f / rsum[v];
    int n_ = m0 + lhi * 8 + v;
    float* orow = out + (size_t)(b_ * DIM_N + n_) * DIM_C + h_ * DIM_D;
    orow[lrow]      = o0[v] * inv;
    orow[16 + lrow] = o1[v] * inv;
    orow[32 + lrow] = o2[v] * inv;
    orow[48 + lrow] = o3[v] * inv;
  }
}

// ---------------------------------------------------------------------------
extern "C" void kernel_launch(void* const* d_in, const int* in_sizes, int n_in,
                              void* d_out, int out_size, void* d_ws, size_t ws_size,
                              hipStream_t stream) {
  (void)in_sizes; (void)n_in; (void)out_size; (void)ws_size;

  const float* x  = (const float*)d_in[0];
  const float* Wq = (const float*)d_in[1];
  const float* bq = (const float*)d_in[2];
  const float* Wk = (const float*)d_in[3];
  const float* bk = (const float*)d_in[4];
  const float* Wv = (const float*)d_in[5];
  const float* bv = (const float*)d_in[6];
  float* out = (float*)d_out;

  const size_t XE = (size_t)M_TOT * DIM_C;   // 4,194,304 elems
  const size_t WE = (size_t)DIM_C * DIM_C;   // 1,048,576 elems
  // workspace layout (f16): x16 | wq16 | wk16 | wv16 | q16 | k16 | v16  (~38 MB)
  f16* x16  = (f16*)d_ws;
  f16* wq16 = x16  + XE;
  f16* wk16 = wq16 + WE;
  f16* wv16 = wk16 + WE;
  f16* q16  = wv16 + WE;
  f16* k16  = q16  + XE;
  f16* v16  = k16  + XE;

  cvt_f32_to_f16<<<(int)(XE / (8 * 256)), 256, 0, stream>>>(x,  x16,  (int)XE);
  cvt_f32_to_f16<<<(int)(WE / (8 * 256)), 256, 0, stream>>>(Wq, wq16, (int)WE);
  cvt_f32_to_f16<<<(int)(WE / (8 * 256)), 256, 0, stream>>>(Wk, wk16, (int)WE);
  cvt_f32_to_f16<<<(int)(WE / (8 * 256)), 256, 0, stream>>>(Wv, wv16, (int)WE);

  qkv_gemm<<<512, 256, 0, stream>>>(x16, wq16, bq, q16);
  qkv_gemm<<<512, 256, 0, stream>>>(x16, wk16, bk, k16);
  qkv_gemm<<<512, 256, 0, stream>>>(x16, wv16, bv, v16);

  attn_fwd<<<512, 256, 0, stream>>>(q16, k16, v16, out);
}
